// ELMo_66142496358588
// MI455X (gfx1250) — compile-verified, matrix operand order checked by reading, not run on hardware
//
#include <hip/hip_runtime.h>
#include <hip/hip_bf16.h>

// ELMo (2-layer BiLSTM + LN) for gfx1250.
// Per-step GEMM [32, K] x [K, 4096] via v_wmma_f32_16x16x32_bf16.
// 64 blocks x 8 wave32: wave = one N-tile (16 cols) x both M-tiles (32 rows),
// K split in half across block groups (partials z1/z2, summed in gate phase).
// A (x_t || h) is staged into LDS with global_load_async_to_lds_b128.

typedef __attribute__((ext_vector_type(16))) __bf16 v16bf;
typedef __attribute__((ext_vector_type(8)))  __bf16 v8bf;
typedef __attribute__((ext_vector_type(8)))  float  v8f;

#define NBLK 64
#define TPB  256

// ---------------- helpers ----------------
__device__ inline float fsig(float x) { return 1.0f / (1.0f + __expf(-x)); }
__device__ inline float ftanh(float x) {
  float e = __expf(2.0f * x);
  return 1.0f - 2.0f / (e + 1.0f);   // safe at +/- inf
}

__device__ inline void grid_barrier(int* bar, int nblk, int& mygen) {
  __threadfence();
  __syncthreads();
  mygen += 1;
  if (threadIdx.x == 0) {
    int arrived = __hip_atomic_fetch_add(&bar[0], 1, __ATOMIC_ACQ_REL, __HIP_MEMORY_SCOPE_AGENT);
    if (arrived == nblk - 1) {
      __hip_atomic_store(&bar[0], 0, __ATOMIC_RELAXED, __HIP_MEMORY_SCOPE_AGENT);
      __hip_atomic_fetch_add(&bar[1], 1, __ATOMIC_RELEASE, __HIP_MEMORY_SCOPE_AGENT);
    } else {
      while (__hip_atomic_load(&bar[1], __ATOMIC_ACQUIRE, __HIP_MEMORY_SCOPE_AGENT) < mygen) {
        __builtin_amdgcn_s_sleep(1);
      }
    }
  }
  __syncthreads();
  __threadfence();
}

__device__ inline v16bf cat16(v8bf a, v8bf b) {
  return __builtin_shufflevector(a, b, 0,1,2,3,4,5,6,7,8,9,10,11,12,13,14,15);
}

// ---------------- embedding gather -> bf16 [B][T][E] ----------------
__global__ void embed_bf16_kernel(const int* __restrict__ tok,
                                  const float* __restrict__ emb,
                                  __bf16* __restrict__ ebf) {
  long i = (long)blockIdx.x * blockDim.x + threadIdx.x; // one v8 per thread
  long base = i * 8;                                    // total 32*256*512 elems
  int bt = (int)(base >> 9);
  int k  = (int)(base & 511);
  int t  = tok[bt];
  const float* src = emb + (long)t * 512 + k;
  v8bf o;
#pragma unroll
  for (int j = 0; j < 8; ++j) o[j] = (__bf16)src[j];
  *(v8bf*)(ebf + base) = o;
}

// ---------------- weight pack: Wt[n][k] = bf16( k<Din ? W[k][n] : U[k-Din][n] ) ----------------
__global__ void wpack_kernel(const float* __restrict__ W, const float* __restrict__ Uw,
                             __bf16* __restrict__ Wt, int Din, int K) {
  long total = (long)4096 * K;
  for (long i = (long)blockIdx.x * blockDim.x + threadIdx.x; i < total;
       i += (long)gridDim.x * blockDim.x) {
    int n = (int)(i % 4096);
    int k = (int)(i / 4096);
    float v = (k < Din) ? W[(long)k * 4096 + n] : Uw[(long)(k - Din) * 4096 + n];
    Wt[(long)n * K + k] = (__bf16)v;
  }
}

// ---------------- state init ----------------
__global__ void init_state_kernel(float* h, float* c, float* o, __bf16* hbf,
                                  int* bar, int zero_hc) {
  int i = blockIdx.x * blockDim.x + threadIdx.x;
  if (i < 2) bar[i] = 0;
  if (i < 32 * 1024) {
    o[i] = 0.0f;
    if (zero_hc) { h[i] = 0.0f; c[i] = 0.0f; hbf[i] = (__bf16)0.0f; }
  }
}

// ---------------- persistent masked-LSTM kernel ----------------
// x(b, step) row base = xbase + b*xbs + sidx*xts, sidx = xrev ? xt0-step : xt0+step
// mask(b, step) = tok[b*256 + (mrev ? mt0-step : mt0+step)] != 0
template <int DIN>
__global__ __launch_bounds__(TPB, 1)
void lstm_kernel(const __bf16* __restrict__ xbase, int xbs, int xts, int xrev, int xt0,
                 const int* __restrict__ tok, int mrev, int mt0,
                 const __bf16* __restrict__ Wt, const float* __restrict__ bias,
                 float* __restrict__ hbuf, float* __restrict__ cbuf,
                 float* __restrict__ obuf, __bf16* __restrict__ hbf,
                 float* __restrict__ z1, float* __restrict__ z2,
                 float* __restrict__ seq, int nsteps, int* __restrict__ bar) {
  constexpr int K   = DIN + 1024;
  constexpr int K2  = K / 2;        // this block's K-slice length
  constexpr int CPR = K2 / 8;       // 16B chunks per A row
  constexpr int NCH = 32 * CPR;     // chunks to stage (multiple of 256)

  __shared__ __align__(16) __bf16 smem[32 * K2];   // A-slice: [32 rows][K2]

  const int lane    = threadIdx.x & 31;
  const int wlocal  = threadIdx.x >> 5;            // 0..7
  const int ksplit  = blockIdx.x >> 5;             // 0 or 1
  const int k0      = ksplit * K2;
  const int nt      = (blockIdx.x & 31) * 8 + wlocal;  // 0..255
  const int lr      = lane & 15;
  const int halfsel = (lane >> 4) * 8;             // ISA 16-bit A/B K-chunking
  const int n       = nt * 16 + lr;                // output column (gate unit)
  const int zr0     = (lane >> 4) * 8;             // first C row in this lane
  float* __restrict__ zs = ksplit ? z2 : z1;
  const __bf16* wrow = Wt + (long)n * K + k0;
  const __bf16* s0 = smem + lr * K2;               // A rows 0-15
  const __bf16* s1 = smem + (16 + lr) * K2;        // A rows 16-31
  const int gtid = blockIdx.x * TPB + threadIdx.x;
  int mygen = 0;

  for (int t = 0; t < nsteps; ++t) {
    const int sidx = xrev ? (xt0 - t) : (xt0 + t);

    // ---------- stage A-slice (x_t || h) into LDS, async ----------
#pragma unroll 4
    for (int c = threadIdx.x; c < NCH; c += TPB) {
      int r  = c / CPR;
      int kk = (c - r * CPR) * 8;
      int k  = k0 + kk;
      const __bf16* src = (k < DIN)
          ? (xbase + (long)r * xbs + (long)sidx * xts + k)
          : (hbf + (long)r * 1024 + (k - DIN));
      unsigned ldsoff = (unsigned)(size_t)(smem + (r * K2 + kk));
      asm volatile("global_load_async_to_lds_b128 %0, %1, off"
                   :: "v"(ldsoff), "v"(src) : "memory");
    }
    asm volatile("s_wait_asynccnt 0x0" ::: "memory");
    __syncthreads();

    // ---------- phase A: partial z = A_slice * Wt_slice^T via WMMA ----------
    v8f acc0, acc1;
#pragma unroll
    for (int r = 0; r < 8; ++r) { acc0[r] = 0.0f; acc1[r] = 0.0f; }

#pragma unroll 4
    for (int kk = 0; kk < K2; kk += 32) {
      v16bf bf = cat16(*(const v8bf*)(wrow + kk + halfsel),
                       *(const v8bf*)(wrow + kk + 16 + halfsel));
      v16bf a0 = cat16(*(const v8bf*)(s0 + kk + halfsel),
                       *(const v8bf*)(s0 + kk + 16 + halfsel));
      acc0 = __builtin_amdgcn_wmma_f32_16x16x32_bf16(false, a0, false, bf,
                                                     (short)0, acc0, false, false);
      v16bf a1 = cat16(*(const v8bf*)(s1 + kk + halfsel),
                       *(const v8bf*)(s1 + kk + 16 + halfsel));
      acc1 = __builtin_amdgcn_wmma_f32_16x16x32_bf16(false, a1, false, bf,
                                                     (short)0, acc1, false, false);
    }
#pragma unroll
    for (int r = 0; r < 8; ++r) {
      zs[(long)(zr0 + r) * 4096 + n]      = acc0[r];
      zs[(long)(16 + zr0 + r) * 4096 + n] = acc1[r];
    }

    grid_barrier(bar, NBLK, mygen);

    // ---------- phase B: gates, state update, mask ----------
    const int midx = mrev ? (mt0 - t) : (mt0 + t);
    for (int idx = gtid; idx < 32 * 1024; idx += NBLK * TPB) {
      int b = idx >> 10, u = idx & 1023;
      bool msk = tok[b * 256 + midx] != 0;
      const float* za = z1 + (long)b * 4096;
      const float* zb = z2 + (long)b * 4096;
      float zi = za[u]        + zb[u]        + bias[u];
      float zf = za[1024 + u] + zb[1024 + u] + bias[1024 + u];
      float zg = za[2048 + u] + zb[2048 + u] + bias[2048 + u];
      float zo = za[3072 + u] + zb[3072 + u] + bias[3072 + u];
      float cold = cbuf[idx], hold = hbuf[idx], oold = obuf[idx];
      float cn = fsig(zf) * cold + fsig(zi) * ftanh(zg);
      float hn = fsig(zo) * ftanh(cn);
      float h2 = msk ? hn : hold;
      float c2 = msk ? cn : cold;
      float ot = msk ? hn : oold;
      hbuf[idx] = h2;
      cbuf[idx] = c2;
      obuf[idx] = ot;
      hbf[idx]  = (__bf16)h2;
      seq[(long)t * 32768 + idx] = ot;   // [t][b][u], step order
    }

    grid_barrier(bar, NBLK, mygen);
  }
}

// ---------------- layernorm over U=1024; one block per (step,batch) row ----------------
__global__ void ln_kernel(const float* __restrict__ seq, const float* __restrict__ g,
                          const float* __restrict__ be, float* __restrict__ outf,
                          __bf16* __restrict__ outbf, int rev) {
  __shared__ float red[8];
  int row = blockIdx.x;            // = s*32 + b
  int s = row >> 5, b = row & 31;
  const float* x = seq + (long)row * 1024;
  float v[4];
  float lsum = 0.0f;
#pragma unroll
  for (int j = 0; j < 4; ++j) { v[j] = x[threadIdx.x + j * 256]; lsum += v[j]; }
#pragma unroll
  for (int off = 16; off; off >>= 1) lsum += __shfl_xor(lsum, off, 32);
  if ((threadIdx.x & 31) == 0) red[threadIdx.x >> 5] = lsum;
  __syncthreads();
  float mean = 0.0f;
#pragma unroll
  for (int i = 0; i < 8; ++i) mean += red[i];
  mean *= (1.0f / 1024.0f);
  __syncthreads();
  float lsq = 0.0f;
#pragma unroll
  for (int j = 0; j < 4; ++j) { float d = v[j] - mean; lsq += d * d; }
#pragma unroll
  for (int off = 16; off; off >>= 1) lsq += __shfl_xor(lsq, off, 32);
  if ((threadIdx.x & 31) == 0) red[threadIdx.x >> 5] = lsq;
  __syncthreads();
  float var = 0.0f;
#pragma unroll
  for (int i = 0; i < 8; ++i) var += red[i];
  var *= (1.0f / 1024.0f);
  float rs = rsqrtf(var + 1e-3f);
#pragma unroll
  for (int j = 0; j < 4; ++j) {
    int u = threadIdx.x + j * 256;
    float y = (v[j] - mean) * rs * g[u] + be[u];
    if (outbf) outbf[(long)row * 1024 + u] = (__bf16)y;
    if (outf) {
      int tt = rev ? (254 - s) : s;
      outf[((long)b * 255 + tt) * 1024 + u] = y;
    }
  }
}

// ---------------- host ----------------
extern "C" void kernel_launch(void* const* d_in, const int* in_sizes, int n_in,
                              void* d_out, int out_size, void* d_ws, size_t ws_size,
                              hipStream_t stream) {
  (void)in_sizes; (void)n_in; (void)out_size; (void)ws_size;
  const int*   x   = (const int*)d_in[0];
  const float* emb = (const float*)d_in[1];
  const float* fW0 = (const float*)d_in[2];
  const float* fU0 = (const float*)d_in[3];
  const float* fb0 = (const float*)d_in[4];
  const float* bW0 = (const float*)d_in[5];
  const float* bU0 = (const float*)d_in[6];
  const float* bb0 = (const float*)d_in[7];
  const float* g0  = (const float*)d_in[8];
  const float* be0 = (const float*)d_in[9];
  const float* fW1 = (const float*)d_in[10];
  const float* fU1 = (const float*)d_in[11];
  const float* fb1 = (const float*)d_in[12];
  const float* bW1 = (const float*)d_in[13];
  const float* bU1 = (const float*)d_in[14];
  const float* bb1 = (const float*)d_in[15];
  const float* g1  = (const float*)d_in[16];
  const float* be1 = (const float*)d_in[17];
  float* out = (float*)d_out;

  char* ws = (char*)d_ws;
  size_t off = 0;
  auto carve = [&](size_t bytes) -> char* {
    off = (off + 255) & ~(size_t)255;
    char* p = ws + off;
    off += bytes;
    return p;
  };
  __bf16* ebf   = (__bf16*)carve((size_t)32 * 256 * 512 * 2);      // 8.4 MB
  __bf16* WtF0  = (__bf16*)carve((size_t)4096 * 1536 * 2);         // 12.6 MB
  __bf16* WtB0  = (__bf16*)carve((size_t)4096 * 1536 * 2);
  __bf16* WtF1  = (__bf16*)carve((size_t)4096 * 2048 * 2);         // 16.8 MB
  __bf16* WtB1  = (__bf16*)carve((size_t)4096 * 2048 * 2);
  float*  seqF  = (float*)carve((size_t)255 * 32 * 1024 * 4);      // 33.4 MB
  float*  seqB  = (float*)carve((size_t)255 * 32 * 1024 * 4);
  __bf16* inFbf = (__bf16*)carve((size_t)255 * 32 * 1024 * 2);     // 16.7 MB
  __bf16* inBbf = (__bf16*)carve((size_t)255 * 32 * 1024 * 2);
  float*  hF    = (float*)carve((size_t)32 * 1024 * 4);
  float*  cF    = (float*)carve((size_t)32 * 1024 * 4);
  float*  oF    = (float*)carve((size_t)32 * 1024 * 4);
  __bf16* hbfF  = (__bf16*)carve((size_t)32 * 1024 * 2);
  float*  hB    = (float*)carve((size_t)32 * 1024 * 4);
  float*  cB    = (float*)carve((size_t)32 * 1024 * 4);
  float*  oB    = (float*)carve((size_t)32 * 1024 * 4);
  __bf16* hbfB  = (__bf16*)carve((size_t)32 * 1024 * 2);
  float*  zF1   = (float*)carve((size_t)32 * 4096 * 4);
  float*  zF2   = (float*)carve((size_t)32 * 4096 * 4);
  float*  zB1   = (float*)carve((size_t)32 * 4096 * 4);
  float*  zB2   = (float*)carve((size_t)32 * 4096 * 4);
  int*    barF  = (int*)carve(256);
  int*    barB  = (int*)carve(256);

  // embeddings -> bf16
  embed_bf16_kernel<<<2048, TPB, 0, stream>>>(x, emb, ebf);
  // pack weights [W;U]^T -> bf16 [4096][K]
  wpack_kernel<<<8192, TPB, 0, stream>>>(fW0, fU0, WtF0, 512, 1536);
  wpack_kernel<<<8192, TPB, 0, stream>>>(bW0, bU0, WtB0, 512, 1536);
  wpack_kernel<<<8192, TPB, 0, stream>>>(fW1, fU1, WtF1, 1024, 2048);
  wpack_kernel<<<8192, TPB, 0, stream>>>(bW1, bU1, WtB1, 1024, 2048);

  // ----- layer 0 -----
  init_state_kernel<<<128, TPB, 0, stream>>>(hF, cF, oF, hbfF, barF, 1);
  init_state_kernel<<<128, TPB, 0, stream>>>(hB, cB, oB, hbfB, barB, 1);
  // forward: x_t = e[:, t], mask[:, t]
  lstm_kernel<512><<<NBLK, TPB, 0, stream>>>(ebf, 256 * 512, 512, 0, 0, x, 0, 0,
                                             WtF0, fb0, hF, cF, oF, hbfF,
                                             zF1, zF2, seqF, 255, barF);
  // backward (step order): x_s = e[:, 255-s], mask[:, 255-s]
  lstm_kernel<512><<<NBLK, TPB, 0, stream>>>(ebf, 256 * 512, 512, 1, 255, x, 1, 255,
                                             WtB0, bb0, hB, cB, oB, hbfB,
                                             zB1, zB2, seqB, 255, barB);
  ln_kernel<<<8160, TPB, 0, stream>>>(seqF, g0, be0, nullptr, inFbf, 0);
  ln_kernel<<<8160, TPB, 0, stream>>>(seqB, g0, be0, nullptr, inBbf, 0);

  // ----- layer 1 (h,c carried; out_prev reset) -----
  init_state_kernel<<<128, TPB, 0, stream>>>(hF, cF, oF, hbfF, barF, 0);
  init_state_kernel<<<128, TPB, 0, stream>>>(hB, cB, oB, hbfB, barB, 0);
  lstm_kernel<1024><<<NBLK, TPB, 0, stream>>>(inFbf, 1024, 32 * 1024, 0, 0, x, 0, 0,
                                              WtF1, fb1, hF, cF, oF, hbfF,
                                              zF1, zF2, seqF, 255, barF);
  // layer-1 backward input already in step order (double reversal cancels)
  lstm_kernel<1024><<<NBLK, TPB, 0, stream>>>(inBbf, 1024, 32 * 1024, 0, 0, x, 1, 255,
                                              WtB1, bb1, hB, cB, oB, hbfB,
                                              zB1, zB2, seqB, 255, barB);
  // final LN -> d_out (fxs then bxs; bxs un-reversed on write)
  ln_kernel<<<8160, TPB, 0, stream>>>(seqF, g1, be1, out, nullptr, 0);
  ln_kernel<<<8160, TPB, 0, stream>>>(seqB, g1, be1, out + (size_t)32 * 255 * 1024,
                                      nullptr, 1);
}
